// FeedForwardMoE_73014444032642
// MI455X (gfx1250) — matrix operand order repaired and checked by reading, not compiled, not run on hardware
//
#include <hip/hip_runtime.h>
#include <hip/hip_bf16.h>
#include <math.h>

// ---------------- problem constants ----------------
constexpr int Bd = 2, Sd = 2048, Dd = 1024, Hd = 4096, Ed = 8, Kd = 2;
constexpr int TOK = Bd * Sd;          // 4096 tokens
constexpr int RAS = TOK * Kd;         // 8192 routed assignments (top-k distinct)

// ---------------- GEMM tiling ----------------
constexpr int BM = 128, BN = 128, BK = 64;      // BK = 2 wmma k-steps of 32
constexpr int MAXRT = TOK / BM;                 // 32 worst-case row tiles / expert

typedef __bf16 bf16;
typedef __attribute__((ext_vector_type(16))) __bf16 v16bf;
typedef __attribute__((ext_vector_type(8)))  float  v8f;

union FragU { uint4 q[2]; v16bf v; };

// ---------------- async LDS copy path (gfx1250) ----------------
#if __has_builtin(__builtin_amdgcn_global_load_async_to_lds_b128) && \
    __has_builtin(__builtin_amdgcn_s_wait_asynccnt)
#define MOE_ASYNC_LDS 1
#else
#define MOE_ASYNC_LDS 0
#endif

#if MOE_ASYNC_LDS
typedef int v4i_vs __attribute__((vector_size(16)));                 // b128 payload type
typedef __attribute__((address_space(1))) v4i_vs* gptr128;           // global (__device__)
typedef __attribute__((address_space(3))) v4i_vs* lptr128;           // LDS (__shared__)
#endif

// copy 64 contiguous bytes global -> LDS (per lane)
__device__ __forceinline__ void copy64_to_lds(const bf16* __restrict__ g, bf16* l) {
#if MOE_ASYNC_LDS
  gptr128 gp = (gptr128)(uintptr_t)g;
  lptr128 lp = (lptr128)(uintptr_t)l;   // low 32 bits of generic LDS ptr = LDS offset
  __builtin_amdgcn_global_load_async_to_lds_b128(gp, lp, 0, 0);
  __builtin_amdgcn_global_load_async_to_lds_b128(gp, lp, 16, 0);
  __builtin_amdgcn_global_load_async_to_lds_b128(gp, lp, 32, 0);
  __builtin_amdgcn_global_load_async_to_lds_b128(gp, lp, 48, 0);
#else
  const uint4* gs = reinterpret_cast<const uint4*>(g);
  uint4* ls = reinterpret_cast<uint4*>(l);
  #pragma unroll
  for (int i = 0; i < 4; ++i) ls[i] = gs[i];
#endif
}

__device__ __forceinline__ void wait_async_copies() {
#if MOE_ASYNC_LDS
  __builtin_amdgcn_s_wait_asynccnt(0);
#endif
}

// ---------------- workspace layout (bytes) ----------------
constexpr size_t WS_COUNTS = 0;                                   // E ints
constexpr size_t WS_OFFS   = 256;                                 // E ints
constexpr size_t WS_TOPKE  = 512;                                 // TOK*2 ints
constexpr size_t WS_TOPKP  = WS_TOPKE + (size_t)TOK * 2 * 4;      // TOK*2 ints
constexpr size_t WS_ROWMAP = WS_TOPKP + (size_t)TOK * 2 * 4;      // RAS ints
constexpr size_t WS_XB     = WS_ROWMAP + (size_t)RAS * 4;         // TOK*D bf16
constexpr size_t WS_W1T    = WS_XB  + (size_t)TOK * Dd * 2;       // E*H*D bf16 (N-major)
constexpr size_t WS_W2T    = WS_W1T + (size_t)Ed * Dd * Hd * 2;   // E*D*H bf16 (N-major)
constexpr size_t WS_HID    = WS_W2T + (size_t)Ed * Dd * Hd * 2;   // RAS*H bf16
constexpr size_t WS_EOUT   = WS_HID + (size_t)RAS * Hd * 2;       // RAS*D f32

// ---------------- small kernels ----------------
__global__ void zero_counts_kernel(int* counts) {
  if (threadIdx.x < Ed) counts[threadIdx.x] = 0;
}

__global__ void cvt_x_kernel(const float* __restrict__ in, bf16* __restrict__ out) {
  size_t n = (size_t)TOK * Dd;
  for (size_t i = (size_t)blockIdx.x * blockDim.x + threadIdx.x; i < n;
       i += (size_t)gridDim.x * blockDim.x)
    out[i] = (bf16)in[i];
}

// [Rr][Cc] f32 -> [Cc][Rr] bf16, per expert slice (blockIdx.z)
__global__ void transpose_cvt_kernel(const float* __restrict__ in, bf16* __restrict__ out,
                                     int Rr, int Cc) {
  __shared__ float tile[32][33];
  const float* src = in  + (size_t)blockIdx.z * Rr * Cc;
  bf16*        dst = out + (size_t)blockIdx.z * Rr * Cc;
  int c0 = blockIdx.x * 32, r0 = blockIdx.y * 32;
  int tx = threadIdx.x, ty = threadIdx.y;            // block (32,8)
  #pragma unroll
  for (int i = 0; i < 32; i += 8)
    tile[ty + i][tx] = src[(size_t)(r0 + ty + i) * Cc + c0 + tx];
  __syncthreads();
  #pragma unroll
  for (int i = 0; i < 32; i += 8)
    dst[(size_t)(c0 + ty + i) * Rr + r0 + tx] = (bf16)tile[tx][ty + i];
}

// one wave32 per token: gate scores, top-2, append to expert lists
__global__ __launch_bounds__(256)
void gate_kernel(const float* __restrict__ x, const float* __restrict__ gw,
                 const float* __restrict__ gb, int* __restrict__ counts,
                 int* __restrict__ topk_e, int* __restrict__ topk_pos) {
  int wave = threadIdx.x >> 5;
  int lane = threadIdx.x & 31;
  int t = blockIdx.x * 8 + wave;
  float acc[Ed] = {};
  const float* xr = x + (size_t)t * Dd;
  for (int d = lane; d < Dd; d += 32) {
    float xv = xr[d];
    const float* g = gw + (size_t)d * Ed;
    #pragma unroll
    for (int e = 0; e < Ed; ++e) acc[e] += xv * g[e];
  }
  #pragma unroll
  for (int e = 0; e < Ed; ++e)
    for (int off = 16; off; off >>= 1) acc[e] += __shfl_xor(acc[e], off, 32);
  if (lane == 0) {
    float s[Ed];
    #pragma unroll
    for (int e = 0; e < Ed; ++e) s[e] = acc[e] + gb[e];
    int e0 = 0;
    for (int e = 1; e < Ed; ++e) if (s[e] > s[e0]) e0 = e;
    int e1 = -1;
    for (int e = 0; e < Ed; ++e)
      if (e != e0 && (e1 < 0 || s[e] > s[e1])) e1 = e;
    int p0 = atomicAdd(&counts[e0], 1);
    int p1 = atomicAdd(&counts[e1], 1);
    topk_e[t * 2] = e0;   topk_e[t * 2 + 1] = e1;
    topk_pos[t * 2] = p0; topk_pos[t * 2 + 1] = p1;
  }
}

__global__ void offsets_kernel(const int* __restrict__ counts, int* __restrict__ offs) {
  if (threadIdx.x == 0) {
    int s = 0;
    for (int e = 0; e < Ed; ++e) { offs[e] = s; s += counts[e]; }
  }
}

__global__ void scatter_kernel(const int* __restrict__ topk_e, const int* __restrict__ topk_pos,
                               const int* __restrict__ offs, int* __restrict__ rowmap) {
  int t = blockIdx.x * blockDim.x + threadIdx.x;
  if (t < TOK) {
    #pragma unroll
    for (int k = 0; k < Kd; ++k)
      rowmap[offs[topk_e[t * 2 + k]] + topk_pos[t * 2 + k]] = t;
  }
}

__global__ void combine_kernel(const float* __restrict__ b2, const float* __restrict__ eout,
                               const int* __restrict__ topk_e, const int* __restrict__ topk_pos,
                               const int* __restrict__ offs, float* __restrict__ out) {
  size_t n = (size_t)TOK * Dd;
  for (size_t i = (size_t)blockIdx.x * blockDim.x + threadIdx.x; i < n;
       i += (size_t)gridDim.x * blockDim.x) {
    int t = (int)(i >> 10);           // D = 1024
    int d = (int)(i & 1023);
    int e0 = topk_e[t * 2], e1 = topk_e[t * 2 + 1];
    size_t r0 = (size_t)offs[e0] + topk_pos[t * 2];
    size_t r1 = (size_t)offs[e1] + topk_pos[t * 2 + 1];
    out[i] = b2[(size_t)e0 * Dd + d] + b2[(size_t)e1 * Dd + d]
           + eout[r0 * Dd + d] + eout[r1 * Dd + d];
  }
}

// ---------------- grouped WMMA GEMM ----------------
// FFN1: A = gathered token rows of xb [cnt x D], B = w1t[e] [H][D] (N-major),
//       epilogue = +b1, exact gelu, store bf16 hid row.
// FFN2: A = hid rows [cnt x H], B = w2t[e] [D][H], store f32 eout row.
// Double-buffered LDS; tile i+1 copied via GLOBAL_LOAD_ASYNC_TO_LDS while
// tile i runs on the WMMA pipe; sync = s_wait_asynccnt 0 + split barrier.
template <bool FFN1>
__global__ __launch_bounds__(256)
void moe_gemm_kernel(const bf16* __restrict__ xb, const bf16* __restrict__ hid_r,
                     bf16* __restrict__ hid_w, const bf16* __restrict__ Wt,
                     const float* __restrict__ b1, float* __restrict__ eout,
                     const int* __restrict__ counts, const int* __restrict__ offs,
                     const int* __restrict__ rowmap) {
  constexpr int Kdim = FFN1 ? Dd : Hd;
  constexpr int Ndim = FFN1 ? Hd : Dd;
  constexpr int NT = Ndim / BN;

  int bx  = blockIdx.x;
  int e   = bx / (MAXRT * NT);
  int rem = bx % (MAXRT * NT);
  int rt  = rem / NT;
  int ntb = rem % NT;
  int cnt = counts[e];
  if (rt * BM >= cnt) return;                 // block-uniform early exit
  int off = offs[e];

  __shared__ __align__(16) bf16 As[2][BM * BK];
  __shared__ __align__(16) bf16 Bs[2][BN * BK];

  int tid   = threadIdx.x;
  int srow  = tid >> 1;                       // staging row 0..127
  int shalf = tid & 1;                        // which 32-element half of BK

  int gr  = rt * BM + srow;
  int grc = min(gr, cnt - 1);
  const bf16* a_src;
  if (FFN1) {
    int tok = rowmap[off + grc];
    a_src = xb + (size_t)tok * Dd;
  } else {
    a_src = hid_r + (size_t)(off + grc) * Hd;
  }
  const bf16* b_src = Wt + ((size_t)e * Ndim + (size_t)(ntb * BN + srow)) * Kdim;

  int lane = tid & 31;
  int wv   = tid >> 5;
  int wm   = wv & 1;                          // 2 wave rows  (64 M each)
  int wn   = wv >> 1;                         // 4 wave cols  (32 N each)
  int g    = lane >> 4;
  int lr   = lane & 15;

  v8f acc[4][2] = {};

  // stage one BM/BN x BK chunk into LDS buffer `buf`
  auto stage = [&](int buf, int kc) {
    copy64_to_lds(a_src + kc + shalf * 32, &As[buf][srow * BK + shalf * 32]);
    copy64_to_lds(b_src + kc + shalf * 32, &Bs[buf][srow * BK + shalf * 32]);
    if (kc + BK < Kdim) {                     // warm GL2 for the chunk after next
      __builtin_prefetch(a_src + kc + BK + shalf * 32, 0, 0);
      __builtin_prefetch(b_src + kc + BK + shalf * 32, 0, 0);
    }
  };

  stage(0, 0);                                // prologue: async copy of chunk 0

  for (int kc = 0, it = 0; kc < Kdim; kc += BK, ++it) {
    int cur = it & 1;
    wait_async_copies();                      // own async stores to LDS done
    __syncthreads();                          // all waves: copies visible, prev reads done
    if (kc + BK < Kdim) stage(cur ^ 1, kc + BK);   // overlap next copy with compute

    #pragma unroll
    for (int ks = 0; ks < BK / 32; ++ks) {
      v16bf aF[4], bF[2];
      #pragma unroll
      for (int mt = 0; mt < 4; ++mt) {
        int m = wm * 64 + mt * 16 + lr;       // A: lanes 0-15 / 16-31 -> k 0-7,16-23 / 8-15,24-31
        FragU u;
        u.q[0] = *reinterpret_cast<const uint4*>(&As[cur][m * BK + ks * 32 + 8 * g]);
        u.q[1] = *reinterpret_cast<const uint4*>(&As[cur][m * BK + ks * 32 + 16 + 8 * g]);
        aF[mt] = u.v;
      }
      #pragma unroll
      for (int nt = 0; nt < 2; ++nt) {
        int n = wn * 32 + nt * 16 + lr;       // B (N-major): lanes 0-15 / 16-31 -> k 0-7,16-23 / 8-15,24-31
        FragU u;
        u.q[0] = *reinterpret_cast<const uint4*>(&Bs[cur][n * BK + ks * 32 + 8 * g]);
        u.q[1] = *reinterpret_cast<const uint4*>(&Bs[cur][n * BK + ks * 32 + 16 + 8 * g]);
        bF[nt] = u.v;
      }
      #pragma unroll
      for (int mt = 0; mt < 4; ++mt)
        #pragma unroll
        for (int nt = 0; nt < 2; ++nt)
          acc[mt][nt] = __builtin_amdgcn_wmma_f32_16x16x32_bf16(
              false, aF[mt], false, bF[nt], (short)0, acc[mt][nt], false, false);
    }
  }

  // ---- epilogue ----
  #pragma unroll
  for (int mt = 0; mt < 4; ++mt) {
    #pragma unroll
    for (int nt = 0; nt < 2; ++nt) {
      #pragma unroll
      for (int r = 0; r < 8; ++r) {
        int ml   = wm * 64 + mt * 16 + r + 8 * g;   // C/D: VGPR r -> M=r / r+8
        int grow = rt * BM + ml;
        if (grow < cnt) {
          int n = ntb * BN + wn * 32 + nt * 16 + lr;
          float v = acc[mt][nt][r];
          if (FFN1) {
            v += b1[(size_t)e * Hd + n];
            v = 0.5f * v * (1.0f + erff(v * 0.70710678118654752f));   // exact gelu
            hid_w[(size_t)(off + grow) * Hd + n] = (bf16)v;
          } else {
            eout[(size_t)(off + grow) * Dd + n] = v;
          }
        }
      }
    }
  }
}

// ---------------- launch ----------------
extern "C" void kernel_launch(void* const* d_in, const int* in_sizes, int n_in,
                              void* d_out, int out_size, void* d_ws, size_t ws_size,
                              hipStream_t stream) {
  const float* x      = (const float*)d_in[0];
  const float* gate_w = (const float*)d_in[1];
  const float* gate_b = (const float*)d_in[2];
  const float* w1     = (const float*)d_in[3];
  const float* b1     = (const float*)d_in[4];
  const float* w2     = (const float*)d_in[5];
  const float* b2     = (const float*)d_in[6];
  float* out = (float*)d_out;

  char* ws = (char*)d_ws;
  int*  counts   = (int*)(ws + WS_COUNTS);
  int*  offs     = (int*)(ws + WS_OFFS);
  int*  topk_e   = (int*)(ws + WS_TOPKE);
  int*  topk_pos = (int*)(ws + WS_TOPKP);
  int*  rowmap   = (int*)(ws + WS_ROWMAP);
  bf16* xb       = (bf16*)(ws + WS_XB);
  bf16* w1t      = (bf16*)(ws + WS_W1T);
  bf16* w2t      = (bf16*)(ws + WS_W2T);
  bf16* hid      = (bf16*)(ws + WS_HID);
  float* eout    = (float*)(ws + WS_EOUT);

  zero_counts_kernel<<<1, 32, 0, stream>>>(counts);

  cvt_x_kernel<<<1024, 256, 0, stream>>>(x, xb);
  // W1: [E][D][H] f32 -> [E][H][D] bf16
  transpose_cvt_kernel<<<dim3(Hd / 32, Dd / 32, Ed), dim3(32, 8), 0, stream>>>(w1, w1t, Dd, Hd);
  // W2: [E][H][D] f32 -> [E][D][H] bf16
  transpose_cvt_kernel<<<dim3(Dd / 32, Hd / 32, Ed), dim3(32, 8), 0, stream>>>(w2, w2t, Hd, Dd);

  gate_kernel<<<TOK / 8, 256, 0, stream>>>(x, gate_w, gate_b, counts, topk_e, topk_pos);
  offsets_kernel<<<1, 32, 0, stream>>>(counts, offs);
  scatter_kernel<<<(TOK + 255) / 256, 256, 0, stream>>>(topk_e, topk_pos, offs, rowmap);

  // FFN1: grouped GEMM [cnt_e x D] x [D x H] -> gelu -> hid
  moe_gemm_kernel<true><<<Ed * MAXRT * (Hd / BN), 256, 0, stream>>>(
      xb, hid, hid, w1t, b1, eout, counts, offs, rowmap);
  // FFN2: grouped GEMM [cnt_e x H] x [H x D] -> eout
  moe_gemm_kernel<false><<<Ed * MAXRT * (Dd / BN), 256, 0, stream>>>(
      xb, hid, hid, w2t, b1, eout, counts, offs, rowmap);

  combine_kernel<<<4096, 256, 0, stream>>>(b2, eout, topk_e, topk_pos, offs, out);
}
